// MyGRU_43525198578433
// MI455X (gfx1250) — matrix operand order repaired and checked by reading, not compiled, not run on hardware
//
#include <hip/hip_runtime.h>
#include <hip/hip_bf16.h>

// ---------------- problem dims (fixed by reference) ----------------
#define T_STEPS 512
#define B_DIM   64
#define I_DIM   512
#define H_DIM   1024
#define G_DIM   (3 * H_DIM)          // 3072
#define KSUM    (I_DIM + H_DIM)      // 1536 concatenated K

// LDS: 3 planes (IH, HHlo, HHhi), each 48 rows x (512 data + 8 pad) bf16.
// Row stride 1040B = 260 dwords == 4 mod 64 -> a wave's ds_load_b128 hits every
// bank exactly twice (optimal). Pad is TDM-generated (interval 256 dw, amount 4 dw).
#define ROW_E    520                  // elements per padded row
#define NROWS    48                   // 16 rows per gate (r,z,n)
#define PLANE_E  (NROWS * ROW_E)      // 24960 elements per plane
#define PLANE_B  (PLANE_E * 2)        // 49920 bytes per plane
#define LDS_BYTES (3 * PLANE_B)       // 149760 bytes (< 320KB WGP LDS)

typedef __attribute__((ext_vector_type(16))) __bf16 v16bf;
typedef __attribute__((ext_vector_type(8)))  float  v8f;

#if defined(__has_builtin)
#if __has_builtin(__builtin_amdgcn_tensor_load_to_lds)
#define USE_TDM 1
#endif
#endif

// ---------------- helpers ----------------
__device__ __forceinline__ unsigned short f32_to_bf16_rne(float f) {
    union { float f; unsigned u; } v; v.f = f;
    unsigned r = 0x7FFFu + ((v.u >> 16) & 1u);
    return (unsigned short)((v.u + r) >> 16);
}

__device__ __forceinline__ float sigmoid_f(float x) {
    return 1.0f / (1.0f + __expf(-x));
}
__device__ __forceinline__ float tanh_f(float x) {
    return 1.0f - 2.0f / (__expf(2.0f * x) + 1.0f);   // saturates to +-1
}

__device__ __forceinline__ v8f wmma_bf16(v16bf a, v16bf b, v8f c) {
    return __builtin_amdgcn_wmma_f32_16x16x32_bf16(
        false, a, false, b, (short)0, c, false, false);
}

#ifdef USE_TDM
typedef __attribute__((ext_vector_type(4))) unsigned tdm_g0_t;
typedef __attribute__((ext_vector_type(8))) int      tdm_g1_t;
typedef __attribute__((ext_vector_type(4))) int      tdm_g2_t;
typedef __attribute__((ext_vector_type(8))) int      tdm_g4_t;  // extra operand (clang-23 6-arg form)

// Issue one TDM load: 3D tile (512 elems x 16 rows x 3 gate-planes) of bf16,
// with LDS padding of 4 dwords after every 256 dwords (one pad per row).
// D# bit packing per CDNA5 ISA 8.3-8.6.
__device__ __forceinline__ void tdm_load_tile(unsigned lds_byte_off,
                                              const unsigned short* gsrc,
                                              unsigned t0stride_e,       // row stride (elems)
                                              unsigned long long t1stride_e, // gate-plane stride (elems)
                                              unsigned tensor_dim0_e) {
    unsigned long long ga = (unsigned long long)(uintptr_t)gsrc;
    tdm_g0_t g0;
    g0[0] = 1u;                                           // count=1, user load
    g0[1] = lds_byte_off;                                 // lds_addr
    g0[2] = (unsigned)(ga & 0xFFFFFFFFu);                 // global_addr[31:0]
    g0[3] = (unsigned)((ga >> 32) & 0x01FFFFFFu) | (2u << 30);  // ga[56:32] | type=2

    tdm_g1_t g1;
    g1[0] = (int)((1u << 16)    // data_size = 2 bytes
                | (1u << 20)    // pad_enable
                | (7u << 22)    // pad_interval = 256 dwords
                | (3u << 25));  // pad_amount  = 4 dwords (16B)
    g1[1] = (int)(tensor_dim0_e << 16);                         // td0[15:0]
    g1[2] = (int)((tensor_dim0_e >> 16) | ((unsigned)G_DIM << 16)); // td0[31:16]|td1[15:0]
    g1[3] = (int)(((unsigned)G_DIM >> 16) | (512u << 16));      // td1[31:16]|tile_dim0=512
    g1[4] = (int)(16u | (3u << 16));                            // tile_dim1=16|tile_dim2=3
    g1[5] = (int)t0stride_e;                                    // dim0_stride[31:0]
    g1[6] = (int)(((unsigned)(t1stride_e & 0xFFFFu)) << 16);    // dim0_stride[47:32]=0 | dim1_stride[15:0]
    g1[7] = (int)(t1stride_e >> 16);                            // dim1_stride[47:16]

    tdm_g2_t g2; g2[0] = 3; g2[1] = 0; g2[2] = 0; g2[3] = 0;    // tensor_dim2=3
    tdm_g2_t g3; g3[0] = 0; g3[1] = 0; g3[2] = 0; g3[3] = 0;
    tdm_g4_t g4; g4[0] = 0; g4[1] = 0; g4[2] = 0; g4[3] = 0;
    g4[4] = 0; g4[5] = 0; g4[6] = 0; g4[7] = 0;

    __builtin_amdgcn_tensor_load_to_lds(g0, g1, g2, g3, g4, 0);
}
#endif

// ---------------- f32 -> bf16 convert ----------------
__global__ void cvt_f32_to_bf16(const float* __restrict__ src,
                                unsigned short* __restrict__ dst, int n) {
    int i = blockIdx.x * blockDim.x + threadIdx.x;
    int stride = gridDim.x * blockDim.x;
    for (; i < n; i += stride) dst[i] = f32_to_bf16_rne(src[i]);
}

// ---------------- barrier state init (ws is poisoned, must zero) ----------------
__global__ void init_barrier(unsigned* bar) {
    if (threadIdx.x < 128) bar[threadIdx.x] = 0u;
}

// ---------------- grid-wide sense barrier ----------------
__device__ __forceinline__ void grid_sync(unsigned* bar) {
    __builtin_amdgcn_fence(__ATOMIC_RELEASE, "agent");
    __syncthreads();
    if (threadIdx.x == 0) {
        unsigned* cnt = bar;
        unsigned* gen = bar + 64;   // separate cacheline
        unsigned g = __hip_atomic_load(gen, __ATOMIC_RELAXED, __HIP_MEMORY_SCOPE_AGENT);
        unsigned a = __hip_atomic_fetch_add(cnt, 1u, __ATOMIC_ACQ_REL, __HIP_MEMORY_SCOPE_AGENT);
        if (a + 1u == gridDim.x) {
            __hip_atomic_store(cnt, 0u, __ATOMIC_RELAXED, __HIP_MEMORY_SCOPE_AGENT);
            __hip_atomic_store(gen, g + 1u, __ATOMIC_RELEASE, __HIP_MEMORY_SCOPE_AGENT);
        } else {
            while (__hip_atomic_load(gen, __ATOMIC_ACQUIRE, __HIP_MEMORY_SCOPE_AGENT) == g) {
                __builtin_amdgcn_s_sleep(2);
            }
        }
    }
    __syncthreads();
    __builtin_amdgcn_fence(__ATOMIC_ACQUIRE, "agent");
}

// ---------------- persistent fused GRU scan ----------------
// grid = 64 blocks (j-slices of H), block = 128 threads (4 waves = m-tiles).
// Per step per wave: r,z over K=1536 (x|h), i_n over K=512, h_n over K=1024
// => 144 v_wmma_f32_16x16x32_bf16. Weights TDM-staged in LDS; h state in VGPRs.
__global__ void __launch_bounds__(128)
gru_scan(const float* __restrict__ h0,
         const float* __restrict__ b_ih, const float* __restrict__ b_hh,
         const unsigned short* __restrict__ xb,   // [T][B][I] bf16
         const unsigned short* __restrict__ wih,  // [3H][I]  bf16
         const unsigned short* __restrict__ whh,  // [3H][H]  bf16
         unsigned short* __restrict__ hbuf,       // [2][B][H] bf16 ping-pong
         float* __restrict__ out,                 // T*B*H outputs, then B*H final h
         unsigned* __restrict__ bar) {
    extern __shared__ unsigned short sW[];        // 3 planes x [48][ROW_E]

    const int tid  = threadIdx.x;
    const int lane = tid & 31;
    const int wave = tid >> 5;       // 0..3 -> m-tile
    const int j0   = blockIdx.x * 16;
    const int m0   = wave * 16;
    const int nl   = lane & 15;
    const int hi   = lane >> 4;

    // ---- stage weights into LDS: plane p, row r -> W row g = (r/16)*H + j0 + r%16,
    //      k-range p==0: W_ih[.,0:512], p==1: W_hh[.,0:512], p==2: W_hh[.,512:1024]
#ifdef USE_TDM
    if (wave == 0) {
        tdm_load_tile(0,           wih + (size_t)j0 * I_DIM,
                      I_DIM, (unsigned long long)H_DIM * I_DIM, I_DIM);
        tdm_load_tile(PLANE_B,     whh + (size_t)j0 * H_DIM,
                      H_DIM, (unsigned long long)H_DIM * H_DIM, H_DIM);
        tdm_load_tile(2 * PLANE_B, whh + (size_t)j0 * H_DIM + 512,
                      H_DIM, (unsigned long long)H_DIM * H_DIM, H_DIM);
        __builtin_amdgcn_s_wait_tensorcnt(0);
    }
#else
    {
        const int CHUNKS = 3 * NROWS * (512 / 8);
        for (int c = tid; c < CHUNKS; c += blockDim.x) {
            int plane = c / (NROWS * 64);
            int rem   = c - plane * (NROWS * 64);
            int row   = rem >> 6;
            int k8    = (rem & 63) * 8;
            int g     = (row >> 4) * H_DIM + j0 + (row & 15);
            const unsigned short* src =
                (plane == 0) ? (wih + (size_t)g * I_DIM + k8)
                             : (whh + (size_t)g * H_DIM + (plane - 1) * 512 + k8);
            *(uint4*)(&sW[plane * PLANE_E + row * ROW_E + k8]) = *(const uint4*)src;
        }
    }
#endif
    __syncthreads();

    // ---- per-lane constants ----
    const int gr = j0 + nl, gz = H_DIM + j0 + nl, gn = 2 * H_DIM + j0 + nl;
    const float brz = b_ih[gr] + b_hh[gr];
    const float bzz = b_ih[gz] + b_hh[gz];
    const float bin = b_ih[gn];
    const float bhn = b_hh[gn];

    // h state in registers: lane holds rows (m0 + v + 8*hi), col (j0+nl)
    float hprev[8];
#pragma unroll
    for (int v = 0; v < 8; ++v)
        hprev[v] = h0[(size_t)(m0 + v + hi * 8) * H_DIM + (j0 + nl)];

    const int arow = m0 + nl;        // A row (batch) for this lane
    const int aoff = hi * 8;         // 16-bit A layout: hi lanes hold K+8 chunk
    const int boff = hi * 8;         // 16-bit B layout: hi lanes hold K+8 chunk
    const int rowR = (0  + nl) * ROW_E;
    const int rowZ = (16 + nl) * ROW_E;
    const int rowN = (32 + nl) * ROW_E;

    auto loadA = [&](const unsigned short* rowp, int kb) -> v16bf {
        union { uint4 q[2]; v16bf v; } f;
        f.q[0] = *(const uint4*)(rowp + kb + aoff);
        f.q[1] = *(const uint4*)(rowp + kb + 16 + aoff);
        return f.v;
    };
    // k is the concatenated K index (0..1535); plane = k>>9 selects IH/HHlo/HHhi
    auto loadB = [&](int rowbase, int k) -> v16bf {
        const unsigned short* p = sW + (k >> 9) * PLANE_E + rowbase + (k & 511) + boff;
        union { uint4 q[2]; v16bf v; } f;
        f.q[0] = *(const uint4*)(p);
        f.q[1] = *(const uint4*)(p + 16);
        return f.v;
    };

    const size_t HN = (size_t)B_DIM * H_DIM;

    for (int t = 0; t < T_STEPS; ++t) {
        const unsigned short* hcur = hbuf + (size_t)(t & 1) * HN;
        unsigned short*       hnxt = hbuf + (size_t)((t + 1) & 1) * HN;
        const unsigned short* xrow = xb + ((size_t)t * B_DIM + arow) * I_DIM;
        const unsigned short* hrow = hcur + (size_t)arow * H_DIM;

        if (t + 1 < T_STEPS)
            __builtin_prefetch(xb + ((size_t)(t + 1) * B_DIM + arow) * I_DIM, 0, 1);

        v8f ar  = {0,0,0,0,0,0,0,0};
        v8f az  = {0,0,0,0,0,0,0,0};
        v8f ain = {0,0,0,0,0,0,0,0};
        v8f ahn = {0,0,0,0,0,0,0,0};

        // x-part: K = 0..511 (feeds r, z, i_n)
#pragma unroll 4
        for (int kk = 0; kk < I_DIM / 32; ++kk) {
            const int kb = kk * 32;
            v16bf A = loadA(xrow, kb);
            ar  = wmma_bf16(A, loadB(rowR, kb), ar);
            az  = wmma_bf16(A, loadB(rowZ, kb), az);
            ain = wmma_bf16(A, loadB(rowN, kb), ain);
        }
        // h-part: K = 512..1535 (feeds r, z, h_n)
#pragma unroll 4
        for (int kk = 0; kk < H_DIM / 32; ++kk) {
            const int kb = kk * 32;
            v16bf A = loadA(hrow, kb);
            const int k = I_DIM + kb;
            ar  = wmma_bf16(A, loadB(rowR, k), ar);
            az  = wmma_bf16(A, loadB(rowZ, k), az);
            ahn = wmma_bf16(A, loadB(rowN, k), ahn);
        }

        // gate math + state update (C/D layout: vgpr v -> row m0+v+8*hi, col j0+nl)
        float* outp = out + (size_t)t * HN;
#pragma unroll
        for (int v = 0; v < 8; ++v) {
            const int b = m0 + v + hi * 8;
            float r = sigmoid_f(ar[v] + brz);
            float z = sigmoid_f(az[v] + bzz);
            float n = tanh_f(ain[v] + bin + r * (ahn[v] + bhn));
            float hn = (1.0f - z) * n + z * hprev[v];
            hprev[v] = hn;
            const size_t o = (size_t)b * H_DIM + (j0 + nl);
            outp[o] = hn;
            hnxt[o] = f32_to_bf16_rne(hn);
        }
        if (t == T_STEPS - 1) {
#pragma unroll
            for (int v = 0; v < 8; ++v) {
                const int b = m0 + v + hi * 8;
                out[(size_t)T_STEPS * HN + (size_t)b * H_DIM + (j0 + nl)] = hprev[v];
            }
        }

        grid_sync(bar);   // next step reads hnxt written by all blocks
    }
}

// ---------------- host launcher ----------------
extern "C" void kernel_launch(void* const* d_in, const int* in_sizes, int n_in,
                              void* d_out, int out_size, void* d_ws, size_t ws_size,
                              hipStream_t stream) {
    const float* x    = (const float*)d_in[0];   // (T,B,I)
    const float* h0   = (const float*)d_in[1];   // (1,B,H)
    const float* W_ih = (const float*)d_in[2];   // (3H,I)
    const float* W_hh = (const float*)d_in[3];   // (3H,H)
    const float* b_ih = (const float*)d_in[4];   // (3H)
    const float* b_hh = (const float*)d_in[5];   // (3H)
    float* out = (float*)d_out;                  // T*B*H outputs ++ B*H final h

    // workspace layout (bf16)
    unsigned short* xb   = (unsigned short*)d_ws;                        // T*B*I
    unsigned short* wih  = xb  + (size_t)T_STEPS * B_DIM * I_DIM;        // 3H*I
    unsigned short* whh  = wih + (size_t)G_DIM * I_DIM;                  // 3H*H
    unsigned short* hbuf = whh + (size_t)G_DIM * H_DIM;                  // 2*B*H ping-pong
    unsigned*       bar  = (unsigned*)(hbuf + 2 * (size_t)B_DIM * H_DIM);

    cvt_f32_to_bf16<<<4096, 256, 0, stream>>>(x,    xb,  T_STEPS * B_DIM * I_DIM);
    cvt_f32_to_bf16<<<1024, 256, 0, stream>>>(W_ih, wih, G_DIM * I_DIM);
    cvt_f32_to_bf16<<<1024, 256, 0, stream>>>(W_hh, whh, G_DIM * H_DIM);
    cvt_f32_to_bf16<<<64,   256, 0, stream>>>(h0,   hbuf, B_DIM * H_DIM); // ping buffer 0
    init_barrier<<<1, 128, 0, stream>>>(bar);

    gru_scan<<<H_DIM / 16, 128, LDS_BYTES, stream>>>(
        h0, b_ih, b_hh, xb, wih, whh, hbuf, out, bar);
}
